// VideoTransformerWithSpeedups_55903294325381
// MI455X (gfx1250) — matrix operand order repaired and checked by reading, not compiled, not run on hardware
//
#include <hip/hip_runtime.h>
#include <hip/hip_bf16.h>
#include <math.h>

typedef __bf16 bf;
typedef __attribute__((ext_vector_type(16))) __bf16 v16bf;
typedef __attribute__((ext_vector_type(8)))  __bf16 v8bf;
typedef __attribute__((ext_vector_type(8)))  float  v8f;

namespace {
constexpr int  kL  = 6;
constexpr int  kD  = 512;
constexpr int  kH  = 8;
constexpr int  kV  = 8192;
constexpr int  kB  = 32;
constexpr int  kT  = 256;
constexpr int  kHD = kD / kH;         // 64
constexpr long kBT = (long)kB * kT;   // 8192 tokens
}

// A-fragment (16x32 bf16 per 16 rows): lane m=lane&15, hi=lane>>4.
// holds K = {p..p+7} u {p+16..p+23} where p = hi*8 (per ISA A layout).
__device__ __forceinline__ v16bf ldA(const bf* p)
{
  v8bf lo = *(const v8bf*)(p);
  v8bf hh = *(const v8bf*)(p + 16);
  return __builtin_shufflevector(lo, hh,
      0, 1, 2, 3, 4, 5, 6, 7, 8, 9, 10, 11, 12, 13, 14, 15);
}

// B-fragment (32x16 bf16): lane n=lane&15 holds K = {p..p+7} u {p+16..p+23}
// where p = hi*8 (same as A layout). Pointer pre-offset by hi*8.
__device__ __forceinline__ v16bf ldB(const bf* p)
{
  v8bf lo = *(const v8bf*)(p);
  v8bf hh = *(const v8bf*)(p + 16);
  return __builtin_shufflevector(lo, hh,
      0, 1, 2, 3, 4, 5, 6, 7, 8, 9, 10, 11, 12, 13, 14, 15);
}

// ---------------------------------------------------------------------------
// Batched NT GEMM: C[M,N] = act( alpha * A[M,K] x B[N,K]^T + bias ),
// bf16 in, f32 out.  One wave -> 32x32 tile (2x2 of 16x16 WMMA, 4 v_wmma per
// K-step of 32).  4 waves/block -> 64x64 block tile.  M,N % 64 == 0, K % 32 == 0.
// Batch offset: z -> (z/batch2)*s?0 + (z%batch2)*s?1
// ---------------------------------------------------------------------------
__global__ __launch_bounds__(128)
void k_gemm_bf16_nt(const bf* __restrict__ A, long sA0, long sA1, int lda,
                    const bf* __restrict__ Bm, long sB0, long sB1, int ldb,
                    const float* __restrict__ bias,
                    float* __restrict__ C, long sC0, long sC1, int ldc,
                    int K, float alpha, int act, int batch2)
{
  const int z  = blockIdx.z;
  const int z0 = z / batch2;
  const int z1 = z - z0 * batch2;
  A  += (long)z0 * sA0 + (long)z1 * sA1;
  Bm += (long)z0 * sB0 + (long)z1 * sB1;
  C  += (long)z0 * sC0 + (long)z1 * sC1;

  const int wave = threadIdx.x >> 5;
  const int lane = threadIdx.x & 31;
  const int m16  = lane & 15;
  const int hi   = lane >> 4;
  const int tM   = blockIdx.y * 64 + (wave >> 1) * 32;
  const int tN   = blockIdx.x * 64 + (wave & 1) * 32;

  v8f acc00 = {0.f,0.f,0.f,0.f,0.f,0.f,0.f,0.f};
  v8f acc01 = acc00, acc10 = acc00, acc11 = acc00;

  const bf* a0 = A  + (long)(tM + m16) * lda + hi * 8;
  const bf* a1 = a0 + (long)16 * lda;
  const bf* b0 = Bm + (long)(tN + m16) * ldb + hi * 8;
  const bf* b1 = b0 + (long)16 * ldb;

  for (int ko = 0; ko < K; ko += 32) {
    const v16bf fa0 = ldA(a0);
    const v16bf fa1 = ldA(a1);
    const v16bf fb0 = ldB(b0);
    const v16bf fb1 = ldB(b1);
    __builtin_prefetch(a0 + 64, 0, 1);   // global_prefetch_b8
    __builtin_prefetch(b0 + 64, 0, 1);
    acc00 = __builtin_amdgcn_wmma_f32_16x16x32_bf16(
        false, fa0, false, fb0, (short)0, acc00, false, false);
    acc01 = __builtin_amdgcn_wmma_f32_16x16x32_bf16(
        false, fa0, false, fb1, (short)0, acc01, false, false);
    acc10 = __builtin_amdgcn_wmma_f32_16x16x32_bf16(
        false, fa1, false, fb0, (short)0, acc10, false, false);
    acc11 = __builtin_amdgcn_wmma_f32_16x16x32_bf16(
        false, fa1, false, fb1, (short)0, acc11, false, false);
    a0 += 32; a1 += 32; b0 += 32; b1 += 32;
  }

  // C layout per 16x16 tile: VGPR r, lanes 0-15 -> M=r, lanes 16-31 -> M=8+r,
  // N = lane&15.
  const int r0 = tM + hi * 8;
  const int c0 = tN + m16;
  const float bb0 = bias ? bias[c0]      : 0.f;
  const float bb1 = bias ? bias[c0 + 16] : 0.f;

  #pragma unroll
  for (int r = 0; r < 8; ++r) {
    float v00 = acc00[r] * alpha + bb0;
    float v01 = acc01[r] * alpha + bb1;
    float v10 = acc10[r] * alpha + bb0;
    float v11 = acc11[r] * alpha + bb1;
    if (act == 1) {
      v00 = 0.5f * v00 * (1.f + erff(v00 * 0.70710678118654752f));
      v01 = 0.5f * v01 * (1.f + erff(v01 * 0.70710678118654752f));
      v10 = 0.5f * v10 * (1.f + erff(v10 * 0.70710678118654752f));
      v11 = 0.5f * v11 * (1.f + erff(v11 * 0.70710678118654752f));
    }
    C[(long)(r0 + r) * ldc + c0]           = v00;
    C[(long)(r0 + r) * ldc + c0 + 16]      = v01;
    C[(long)(r0 + 16 + r) * ldc + c0]      = v10;
    C[(long)(r0 + 16 + r) * ldc + c0 + 16] = v11;
  }
}

// ------------------------------ elementwise --------------------------------
__global__ void k_f32_to_bf16(const float* __restrict__ in,
                              bf* __restrict__ out, long n)
{
  long i = (long)blockIdx.x * 256 + threadIdx.x;
  if (i < n) out[i] = (bf)in[i];
}

__global__ void k_add(float* __restrict__ x, const float* __restrict__ y, long n)
{
  long i = (long)blockIdx.x * 256 + threadIdx.x;
  if (i < n) x[i] += y[i];
}

__global__ void k_embed(const int* __restrict__ tok,
                        const float* __restrict__ te,
                        const float* __restrict__ pe,
                        float* __restrict__ x)
{
  const long i  = (long)blockIdx.x * 256 + threadIdx.x;   // over kBT*kD
  const long bt = i >> 9;                                 // / 512
  const int  d  = (int)(i & 511);
  const int  t  = (int)(bt & (kT - 1));
  x[i] = te[(long)tok[bt] * kD + d] + pe[(long)t * kD + d];
}

// f32 [B,T,ldin] (column slice [coloff, coloff+kD)) -> bf16 [B,kD,T]
__global__ void k_transpose_bf(const float* __restrict__ in,
                               bf* __restrict__ out, int ldin, int coloff)
{
  long i = (long)blockIdx.x * 256 + threadIdx.x;          // over B*D*T
  if (i >= (long)kB * kD * kT) return;
  int  t   = (int)(i % kT);
  long rem = i / kT;
  int  d   = (int)(rem % kD);
  int  b   = (int)(rem / kD);
  out[i] = (bf)in[((long)b * kT + t) * ldin + coloff + d];
}

// LayerNorm over rows of length 512; one 256-thread block per row.
__global__ __launch_bounds__(256)
void k_layernorm(const float* __restrict__ x, const float* __restrict__ g,
                 const float* __restrict__ b, float* __restrict__ y)
{
  const long base = (long)blockIdx.x * kD;
  const int  t    = threadIdx.x;
  float v0 = x[base + t], v1 = x[base + t + 256];
  __shared__ float s1[256], s2[256];
  s1[t] = v0 + v1;
  s2[t] = v0 * v0 + v1 * v1;
  __syncthreads();
  for (int o = 128; o > 0; o >>= 1) {
    if (t < o) { s1[t] += s1[t + o]; s2[t] += s2[t + o]; }
    __syncthreads();
  }
  const float mean = s1[0] * (1.f / kD);
  const float var  = s2[0] * (1.f / kD) - mean * mean;
  const float inv  = rsqrtf(var + 1e-5f);
  y[base + t]       = (v0 - mean) * inv * g[t]       + b[t];
  y[base + t + 256] = (v1 - mean) * inv * g[t + 256] + b[t + 256];
}

// Softmax over rows of length 256 (in place); one block per row.
__global__ __launch_bounds__(256)
void k_softmax256(float* __restrict__ x)
{
  const long base = (long)blockIdx.x * 256;
  const int  t    = threadIdx.x;
  float v = x[base + t];
  __shared__ float red[256];
  red[t] = v;
  __syncthreads();
  for (int o = 128; o > 0; o >>= 1) {
    if (t < o) red[t] = fmaxf(red[t], red[t + o]);
    __syncthreads();
  }
  const float mx = red[0];
  __syncthreads();
  const float e = __expf(v - mx);
  red[t] = e;
  __syncthreads();
  for (int o = 128; o > 0; o >>= 1) {
    if (t < o) red[t] += red[t + o];
    __syncthreads();
  }
  x[base + t] = e / red[0];
}

// fmean[b,d] = mean_t f[b,t,d]
__global__ void k_colmean(const float* __restrict__ f, float* __restrict__ out)
{
  int i = blockIdx.x * 256 + threadIdx.x;   // over B*D
  if (i >= kB * kD) return;
  int b = i / kD, d = i - b * kD;
  const float* p = f + (long)b * kT * kD + d;
  float s = 0.f;
  for (int t = 0; t < kT; ++t) s += p[(long)t * kD];
  out[i] = s * (1.f / kT);
}

// reg[b] = softplus( relu(fmean[b] @ rw1^T + rb1) @ rw2^T + rb2 )
__global__ __launch_bounds__(64)
void k_reg(const float* __restrict__ fmean, const float* __restrict__ rw1,
           const float* __restrict__ rb1,   const float* __restrict__ rw2,
           const float* __restrict__ rb2,   float* __restrict__ reg)
{
  const int b = blockIdx.x, j = threadIdx.x;
  const float* fm = fmean + (long)b * kD;
  float acc = rb1[j];
  for (int d = 0; d < kD; ++d) acc += fm[d] * rw1[(long)j * kD + d];
  float h = fmaxf(acc, 0.f) * rw2[j];
  __shared__ float red[64];
  red[j] = h;
  __syncthreads();
  for (int o = 32; o > 0; o >>= 1) {
    if (j < o) red[j] += red[j + o];
    __syncthreads();
  }
  if (j == 0) {
    float zv = red[0] + rb2[0];
    reg[b] = (zv > 20.f) ? zv : log1pf(__expf(zv));
  }
}

__global__ void k_add_eye(float* __restrict__ A, const float* __restrict__ reg)
{
  int i = blockIdx.x * 256 + threadIdx.x;   // over B*D
  if (i >= kB * kD) return;
  int b = i / kD, d = i - b * kD;
  A[(long)b * kD * kD + (long)d * kD + d] += reg[b];
}

// One fused CG step per [b,n] row of length D=512; 256 threads, 2 elems each.
__global__ __launch_bounds__(256)
void k_cg_step(const float* __restrict__ Ap, float* __restrict__ xs,
               float* __restrict__ r, float* __restrict__ pv)
{
  const long base = (long)blockIdx.x * kD;
  const int  t    = threadIdx.x;
  const int  d0 = t, d1 = t + 256;
  float r0 = r[base + d0],  r1 = r[base + d1];
  float p0 = pv[base + d0], p1 = pv[base + d1];
  float a0 = Ap[base + d0], a1 = Ap[base + d1];
  __shared__ float s1[256], s2[256];
  s1[t] = r0 * r0 + r1 * r1;
  s2[t] = p0 * a0 + p1 * a1;
  __syncthreads();
  for (int o = 128; o > 0; o >>= 1) {
    if (t < o) { s1[t] += s1[t + o]; s2[t] += s2[t + o]; }
    __syncthreads();
  }
  const float rns = s1[0], pAp = s2[0];
  __syncthreads();
  const float alpha = rns / (pAp + 1e-8f);
  xs[base + d0] += alpha * p0;
  xs[base + d1] += alpha * p1;
  const float rn0 = r0 - alpha * a0, rn1 = r1 - alpha * a1;
  s1[t] = rn0 * rn0 + rn1 * rn1;
  __syncthreads();
  for (int o = 128; o > 0; o >>= 1) {
    if (t < o) s1[t] += s1[t + o];
    __syncthreads();
  }
  const float beta = s1[0] / (rns + 1e-8f);
  pv[base + d0] = rn0 + beta * p0;
  pv[base + d1] = rn1 + beta * p1;
  r[base + d0] = rn0;
  r[base + d1] = rn1;
}

// ------------------------------- host side ---------------------------------
static void to_bf16(hipStream_t st, const float* in, bf* out, long n)
{
  k_f32_to_bf16<<<dim3((unsigned)((n + 255) / 256)), dim3(256), 0, st>>>(in, out, n);
}

static void addv(hipStream_t st, float* x, const float* y, long n)
{
  k_add<<<dim3((unsigned)((n + 255) / 256)), dim3(256), 0, st>>>(x, y, n);
}

static void ln(hipStream_t st, const float* x, const float* g, const float* b,
               float* y, long rows)
{
  k_layernorm<<<dim3((unsigned)rows), dim3(256), 0, st>>>(x, g, b, y);
}

static void transpose_bf(hipStream_t st, const float* in, bf* out,
                         int ldin, int coloff)
{
  k_transpose_bf<<<dim3((unsigned)(kBT * kD / 256)), dim3(256), 0, st>>>(
      in, out, ldin, coloff);
}

static void gemm(hipStream_t st,
                 const bf* A, long sA0, long sA1, int lda,
                 const bf* B, long sB0, long sB1, int ldb,
                 const float* bias,
                 float* C, long sC0, long sC1, int ldc,
                 int M, int N, int K, int nbatch, int batch2,
                 float alpha, int act)
{
  dim3 g(N / 64, M / 64, nbatch), blk(128, 1, 1);
  k_gemm_bf16_nt<<<g, blk, 0, st>>>(A, sA0, sA1, lda, B, sB0, sB1, ldb, bias,
                                    C, sC0, sC1, ldc, K, alpha, act, batch2);
}

extern "C" void kernel_launch(void* const* d_in, const int* in_sizes, int n_in,
                              void* d_out, int out_size, void* d_ws, size_t ws_size,
                              hipStream_t stream)
{
  (void)in_sizes; (void)n_in; (void)out_size; (void)ws_size;
  const int*   tokens = (const int*)  d_in[0];
  const float* tembed = (const float*)d_in[1];
  const float* pembed = (const float*)d_in[2];
  const float* n1g = (const float*)d_in[3];
  const float* n1b = (const float*)d_in[4];
  const float* qkvw = (const float*)d_in[5];
  const float* ow   = (const float*)d_in[6];
  const float* ob   = (const float*)d_in[7];
  const float* n2g = (const float*)d_in[8];
  const float* n2b = (const float*)d_in[9];
  const float* mw1 = (const float*)d_in[10];
  const float* mb1 = (const float*)d_in[11];
  const float* mw2 = (const float*)d_in[12];
  const float* mb2 = (const float*)d_in[13];
  const float* mlg = (const float*)d_in[14];
  const float* mlb = (const float*)d_in[15];
  const float* rw1 = (const float*)d_in[16];
  const float* rb1 = (const float*)d_in[17];
  const float* rw2 = (const float*)d_in[18];
  const float* rb2 = (const float*)d_in[19];
  const float* n3g = (const float*)d_in[20];
  const float* n3b = (const float*)d_in[21];
  const float* pw1 = (const float*)d_in[22];
  const float* pb1 = (const float*)d_in[23];
  const float* pw2 = (const float*)d_in[24];
  const float* pb2 = (const float*)d_in[25];
  const float* nog = (const float*)d_in[26];
  const float* nob = (const float*)d_in[27];
  const float* hw  = (const float*)d_in[28];
  const float* hb  = (const float*)d_in[29];

  // ---- workspace carve-up ----
  char* p = (char*)d_ws;
  auto carve = [&](size_t bytes) -> char* {
    char* r = p; p += (bytes + 255) & ~(size_t)255; return r;
  };
  float* xbuf   = (float*)carve((size_t)kBT * kD * 4);
  float* hbuf   = (float*)carve((size_t)kBT * kD * 4);
  bf*    gbf    = (bf*)   carve((size_t)kBT * kD * 2);          // generic bf16 [BT,D]
  float* qkv    = (float*)carve((size_t)kBT * 3 * kD * 4);
  bf*    qkv_bf = (bf*)   carve((size_t)kBT * 3 * kD * 2);
  bf*    vT_bf  = (bf*)   carve((size_t)kBT * kD * 2);          // [B, D, T] of v
  float* big    = (float*)carve((size_t)kBT * 4 * kD * 4);      // scores / hidden
  bf*    big_bf = (bf*)   carve((size_t)kBT * 4 * kD * 2);
  float* ao     = (float*)carve((size_t)kBT * kD * 4);          // attn heads out / mesa_out
  float* aout   = (float*)carve((size_t)kBT * kD * 4);          // attn_out / mlp_out
  float* fbuf   = (float*)carve((size_t)kBT * kD * 4);
  bf*    f_bf   = (bf*)   carve((size_t)kBT * kD * 2);
  bf*    fT_bf  = (bf*)   carve((size_t)kBT * kD * 2);
  bf*    aoT_bf = (bf*)   carve((size_t)kBT * kD * 2);
  float* Amat   = (float*)carve((size_t)kB * kD * kD * 4);
  bf*    A_bf   = (bf*)   carve((size_t)kB * kD * kD * 2);
  bf*    pv_bf  = (bf*)   carve((size_t)kB * kD * kD * 2);      // also xs_bf
  float* fmean  = (float*)carve((size_t)kB * kD * 4);
  float* regv   = (float*)carve((size_t)kB * 4);
  bf* w_qkv = (bf*)carve((size_t)kL * 3 * kD * kD * 2);
  bf* w_ow  = (bf*)carve((size_t)kL * kD * kD * 2);
  bf* w_m1  = (bf*)carve((size_t)kL * 2 * kD * kD * 2);
  bf* w_m2  = (bf*)carve((size_t)kL * 2 * kD * kD * 2);
  bf* w_p1  = (bf*)carve((size_t)kL * 4 * kD * kD * 2);
  bf* w_p2  = (bf*)carve((size_t)kL * 4 * kD * kD * 2);
  bf* w_hd  = (bf*)carve((size_t)kV * kD * 2);

  // CG scratch lives in d_out (fully dead before final head GEMM writes it)
  const size_t nBDD = (size_t)kB * kD * kD;
  float* bmAp = (float*)d_out;
  float* xs   = bmAp + nBDD;
  float* rr   = xs + nBDD;
  float* pvv  = rr + nBDD;

  // ---- weight conversion (deterministic, redone every call) ----
  to_bf16(stream, qkvw, w_qkv, (long)kL * 3 * kD * kD);
  to_bf16(stream, ow,   w_ow,  (long)kL * kD * kD);
  to_bf16(stream, mw1,  w_m1,  (long)kL * 2 * kD * kD);
  to_bf16(stream, mw2,  w_m2,  (long)kL * 2 * kD * kD);
  to_bf16(stream, pw1,  w_p1,  (long)kL * 4 * kD * kD);
  to_bf16(stream, pw2,  w_p2,  (long)kL * 4 * kD * kD);
  to_bf16(stream, hw,   w_hd,  (long)kV * kD);

  // ---- embedding ----
  k_embed<<<dim3((unsigned)(kBT * kD / 256)), dim3(256), 0, stream>>>(
      tokens, tembed, pembed, xbuf);

  for (int l = 0; l < kL; ++l) {
    const bf* Wqkv = w_qkv + (size_t)l * 3 * kD * kD;
    const bf* Wow  = w_ow  + (size_t)l * kD * kD;
    const bf* Wm1  = w_m1  + (size_t)l * 2 * kD * kD;
    const bf* Wm2  = w_m2  + (size_t)l * 2 * kD * kD;
    const bf* Wp1  = w_p1  + (size_t)l * 4 * kD * kD;
    const bf* Wp2  = w_p2  + (size_t)l * 4 * kD * kD;

    // ---------------- attention ----------------
    ln(stream, xbuf, n1g + l * kD, n1b + l * kD, hbuf, kBT);
    to_bf16(stream, hbuf, gbf, kBT * kD);
    gemm(stream, gbf, 0, 0, kD, Wqkv, 0, 0, kD, nullptr,
         qkv, 0, 0, 3 * kD, (int)kBT, 3 * kD, kD, 1, 1, 1.f, 0);
    to_bf16(stream, qkv, qkv_bf, kBT * 3 * kD);
    transpose_bf(stream, qkv, vT_bf, 3 * kD, 2 * kD);   // vT [B,D,T]

    // scores[b,h] = q @ k^T * HD^-0.5   (M=N=T, K=HD)
    gemm(stream,
         qkv_bf,        (long)kT * 3 * kD, kHD, 3 * kD,
         qkv_bf + kD,   (long)kT * 3 * kD, kHD, 3 * kD,
         nullptr,
         big, (long)kH * kT * kT, (long)kT * kT, kT,
         kT, kT, kHD, kB * kH, kH, 0.125f, 0);
    k_softmax256<<<dim3(kB * kH * kT), dim3(256), 0, stream>>>(big);
    to_bf16(stream, big, big_bf, (long)kB * kH * kT * kT);

    // ao[b,h] = attn @ v = attn @ (vT)^T   (all-NT now), merged into [B,T,D]
    gemm(stream,
         big_bf, (long)kH * kT * kT, (long)kT * kT, kT,
         vT_bf,  (long)kD * kT,      (long)kHD * kT, kT,
         nullptr,
         ao, (long)kT * kD, kHD, kD,
         kT, kHD, kT, kB * kH, kH, 1.f, 0);

    to_bf16(stream, ao, gbf, kBT * kD);
    gemm(stream, gbf, 0, 0, kD, Wow, 0, 0, kD, ob + l * kD,
         aout, 0, 0, kD, (int)kBT, kD, kD, 1, 1, 1.f, 0);
    addv(stream, xbuf, aout, kBT * kD);

    // ---------------- MesaNet ----------------
    ln(stream, xbuf, n2g + l * kD, n2b + l * kD, hbuf, kBT);
    to_bf16(stream, hbuf, gbf, kBT * kD);
    gemm(stream, gbf, 0, 0, kD, Wm1, 0, 0, kD, mb1 + l * 2 * kD,
         big, 0, 0, 2 * kD, (int)kBT, 2 * kD, kD, 1, 1, 1.f, 1);   // gelu
    to_bf16(stream, big, big_bf, kBT * 2 * kD);
    gemm(stream, big_bf, 0, 0, 2 * kD, Wm2, 0, 0, 2 * kD, mb2 + l * kD,
         hbuf, 0, 0, kD, (int)kBT, kD, 2 * kD, 1, 1, 1.f, 0);
    ln(stream, hbuf, mlg + l * kD, mlb + l * kD, fbuf, kBT);

    to_bf16(stream, fbuf, f_bf, kBT * kD);
    transpose_bf(stream, fbuf, fT_bf, kD, 0);
    transpose_bf(stream, aout, aoT_bf, kD, 0);

    // A[b] = f^T f   (M=N=D, K=T)
    gemm(stream, fT_bf, (long)kD * kT, 0, kT, fT_bf, (long)kD * kT, 0, kT,
         nullptr, Amat, (long)kD * kD, 0, kD, kD, kD, kT, kB, 1, 1.f, 0);

    // reg net + A += reg*I
    k_colmean<<<dim3(kB * kD / 256), dim3(256), 0, stream>>>(fbuf, fmean);
    k_reg<<<dim3(kB), dim3(64), 0, stream>>>(fmean, rw1 + (size_t)l * 64 * kD,
                                             rb1 + l * 64, rw2 + l * 64,
                                             rb2 + l, regv);
    k_add_eye<<<dim3(kB * kD / 256), dim3(256), 0, stream>>>(Amat, regv);
    to_bf16(stream, Amat, A_bf, (long)nBDD);

    // bm[b] = ao^T f   -> initial r = pv = bm, xs = 0
    gemm(stream, aoT_bf, (long)kD * kT, 0, kT, fT_bf, (long)kD * kT, 0, kT,
         nullptr, bmAp, (long)kD * kD, 0, kD, kD, kD, kT, kB, 1, 1.f, 0);
    hipMemcpyAsync(rr,  bmAp, nBDD * 4, hipMemcpyDeviceToDevice, stream);
    hipMemcpyAsync(pvv, bmAp, nBDD * 4, hipMemcpyDeviceToDevice, stream);
    hipMemsetAsync(xs, 0, nBDD * 4, stream);

    for (int it = 0; it < 4; ++it) {
      to_bf16(stream, pvv, pv_bf, (long)nBDD);
      // Ap[b,n,d] = sum_e pv[n,e] * A[d,e]  (NT form matches mv())
      gemm(stream, pv_bf, (long)kD * kD, 0, kD, A_bf, (long)kD * kD, 0, kD,
           nullptr, bmAp, (long)kD * kD, 0, kD, kD, kD, kD, kB, 1, 1.f, 0);
      k_cg_step<<<dim3(kB * kD), dim3(256), 0, stream>>>(bmAp, xs, rr, pvv);
    }

    // mesa_out[b] = f @ xs^T
    to_bf16(stream, xs, pv_bf, (long)nBDD);
    gemm(stream, f_bf, (long)kT * kD, 0, kD, pv_bf, (long)kD * kD, 0, kD,
         nullptr, ao, (long)kT * kD, 0, kD, kT, kD, kD, kB, 1, 1.f, 0);
    addv(stream, xbuf, ao, kBT * kD);

    // ---------------- MLP ----------------
    ln(stream, xbuf, n3g + l * kD, n3b + l * kD, hbuf, kBT);
    to_bf16(stream, hbuf, gbf, kBT * kD);
    gemm(stream, gbf, 0, 0, kD, Wp1, 0, 0, kD, pb1 + l * 4 * kD,
         big, 0, 0, 4 * kD, (int)kBT, 4 * kD, kD, 1, 1, 1.f, 1);   // gelu
    to_bf16(stream, big, big_bf, kBT * 4 * kD);
    gemm(stream, big_bf, 0, 0, 4 * kD, Wp2, 0, 0, 4 * kD, pb2 + l * kD,
         aout, 0, 0, kD, (int)kBT, kD, 4 * kD, 1, 1, 1.f, 0);
    addv(stream, xbuf, aout, kBT * kD);
  }

  // ---- final LN + head ----
  ln(stream, xbuf, nog, nob, hbuf, kBT);
  to_bf16(stream, hbuf, gbf, kBT * kD);
  gemm(stream, gbf, 0, 0, kD, w_hd, 0, 0, kD, hb,
       (float*)d_out, 0, 0, kV, (int)kBT, kV, kD, 1, 1, 1.f, 0);
}
